// Second_Order_CAM_Module_77653008711847
// MI455X (gfx1250) — matrix operand order repaired and checked by reading, not compiled, not run on hardware
//
#include <hip/hip_runtime.h>
#include <hip/hip_bf16.h>
#include <math.h>

// ---------------------------------------------------------------------------
// Second-order channel attention (CAM) for x:(16,512,64,64) f32.
//   sigma = (Xf@Xf^T - N*mu*mu^T) * 1/(N*sqrt(N))   [symmetric: upper tiles]
//   attn  = softmax(sigma)
//   d_out = gamma * (attn @ Xf) + x
// GEMMs on v_wmma_f32_16x16x32_bf16 (f32 accum), LDS-staged bf16 tiles,
// 128x128 block tile, 4 waves (2x2), 64x64 per wave (4x4 wmma frags).
// Global/LDS addressing uses one base pointer per tile with constant
// per-iteration offsets (folds into IOFFSET; avoids per-iteration u64 adds).
// ---------------------------------------------------------------------------

typedef __bf16 bf16_t;
typedef __attribute__((ext_vector_type(16))) __bf16 v16bf;
typedef __attribute__((ext_vector_type(8)))  float  v8f;

constexpr int   Bn = 16, Cn = 512, Hn = 64, Wn = 64;
constexpr int   Nn = Hn * Wn;                            // 4096
constexpr float SIGMA_SCALE = 1.0f / (4096.0f * 64.0f);  // 1/(N*sqrt(N))

// ------------------------- WMMA fragment gathers ---------------------------
// A 16x32 bf16 (MxK), row-major LDS tile:
// lanes 0-15: M=lane, K 0-7 in v0-3 (2/VGPR), K 16-23 in v4-7
// lanes 16-31: M=lane-16, K 8-15 in v0-3, K 24-31 in v4-7
__device__ __forceinline__ v16bf frag_a_ld(const bf16_t* base, int ldk) {
  const int lane = threadIdx.x & 31;
  const int hl   = lane >> 4;
  const int m    = lane & 15;
  v16bf f;
#pragma unroll
  for (int j = 0; j < 8; ++j) {
    const int k = (j < 4) ? (8 * hl + 2 * j) : (16 + 8 * hl + 2 * (j - 4));
    const bf16_t* p = base + m * ldk + k;
    f[2 * j + 0] = p[0];
    f[2 * j + 1] = p[1];
  }
  return f;
}

// B 32x16 bf16 (KxN), LDS tile stored as [n][k] (K contiguous):
// lanes 0-15: N=lane, K 0-7 in v0-3, K 16-23 in v4-7; lanes 16-31: N=lane-16, K 8-15, 24-31
__device__ __forceinline__ v16bf frag_b_ld(const bf16_t* base, int ldk) {
  const int lane = threadIdx.x & 31;
  const int hl   = lane >> 4;
  const int n    = lane & 15;
  v16bf f;
#pragma unroll
  for (int i = 0; i < 16; ++i)
    f[i] = base[n * ldk + ((i < 8) ? (8 * hl + i) : (16 + 8 * hl + (i - 8)))];
  return f;
}

// ------------------------- kernel 1: row means -----------------------------
__global__ __launch_bounds__(256)
void cam_row_mean(const float* __restrict__ x, float* __restrict__ mean) {
  const int row  = blockIdx.x;                 // 0 .. B*C-1
  const float* p = x + (size_t)row * Nn;
  float s = 0.f;
  for (int i = threadIdx.x; i < Nn; i += 256) s += p[i];
#pragma unroll
  for (int off = 16; off; off >>= 1) s += __shfl_xor(s, off, 32);
  __shared__ float red[8];
  if ((threadIdx.x & 31) == 0) red[threadIdx.x >> 5] = s;
  __syncthreads();
  if (threadIdx.x == 0) {
    float t = 0.f;
#pragma unroll
    for (int i = 0; i < 8; ++i) t += red[i];
    mean[row] = t * (1.0f / (float)Nn);
  }
}

// ------ kernel 2: sigma = (Gram - N*mu*mu^T)*scale, upper tiles only -------
// Per batch: symmetric 512x512, K=4096. blockIdx.x enumerates the 10 upper-
// triangular 128x128 tile pairs; off-diagonal results are mirrored on store.
// Diagonal tiles reuse sA as the B operand (skip half the global traffic).
__global__ __launch_bounds__(128)
void cam_sigma_gemm(const float* __restrict__ x, const float* __restrict__ mean,
                    float* __restrict__ sigma) {
  constexpr int TM = 128, TN = 128, TK = 64, LDK = TK + 8;
  __shared__ bf16_t sA[TM * LDK];   // Xf tile rows c0..  [c][k]
  __shared__ bf16_t sB[TN * LDK];   // Xf tile rows d0..  [d][k]

  const int b = blockIdx.z;
  int p = blockIdx.x, ti = 0;                       // linear -> (ti, tj), ti<=tj
  while (p >= (Cn / TM) - ti) { p -= (Cn / TM) - ti; ++ti; }
  const int tj   = ti + p;
  const int c0   = ti * TM;
  const int d0   = tj * TN;
  const bool diag = (ti == tj);

  const float* xb = x + (size_t)b * Cn * Nn;
  const int t    = threadIdx.x;
  const int wave = t >> 5;
  const int wm   = wave >> 1;        // 0..1
  const int wn   = wave & 1;         // 0..1

  // per-thread load bases: iteration offsets are compile-time constants
  const int rbase = t >> 4;          // 0..7  (row within tile, step 8 per iter)
  const int cbase = (t & 15) << 2;   // 0..60 (float4 column)
  const float* pA = xb + (size_t)(c0 + rbase) * Nn + cbase;
  const float* pB = xb + (size_t)(d0 + rbase) * Nn + cbase;
  bf16_t* dA = &sA[rbase * LDK + cbase];
  bf16_t* dB = &sB[rbase * LDK + cbase];
  const bf16_t* sBsel = diag ? sA : sB;      // B-operand tile (uniform)

  v8f acc[4][4] = {};

  for (int k0 = 0; k0 < Nn; k0 += TK) {
    const bool pf = (k0 + TK < Nn);
#pragma unroll
    for (int i = 0; i < 16; ++i) {           // 8 rows per iter step
      const float4 va = *(const float4*)(pA + i * 8 * Nn);
      bf16_t* da = dA + i * 8 * LDK;
      da[0] = (bf16_t)va.x; da[1] = (bf16_t)va.y;
      da[2] = (bf16_t)va.z; da[3] = (bf16_t)va.w;
      if (pf) __builtin_prefetch((const void*)(pA + i * 8 * Nn + TK), 0, 1);
    }
    if (!diag) {
#pragma unroll
      for (int i = 0; i < 16; ++i) {
        const float4 vb = *(const float4*)(pB + i * 8 * Nn);
        bf16_t* db = dB + i * 8 * LDK;
        db[0] = (bf16_t)vb.x; db[1] = (bf16_t)vb.y;
        db[2] = (bf16_t)vb.z; db[3] = (bf16_t)vb.w;
        if (pf) __builtin_prefetch((const void*)(pB + i * 8 * Nn + TK), 0, 1);
      }
    }
    __syncthreads();

#pragma unroll
    for (int kk = 0; kk < TK; kk += 32) {
      v16bf af[4], bfr[4];
#pragma unroll
      for (int mi = 0; mi < 4; ++mi)
        af[mi] = frag_a_ld(&sA[(wm * 64 + mi * 16) * LDK + kk], LDK);
#pragma unroll
      for (int ni = 0; ni < 4; ++ni)
        bfr[ni] = frag_b_ld(&sBsel[(wn * 64 + ni * 16) * LDK + kk], LDK);
#pragma unroll
      for (int mi = 0; mi < 4; ++mi)
#pragma unroll
        for (int ni = 0; ni < 4; ++ni)
          acc[mi][ni] = __builtin_amdgcn_wmma_f32_16x16x32_bf16(
              false, af[mi], false, bfr[ni], (short)0, acc[mi][ni], false, false);
    }
    __syncthreads();
    pA += TK;
    pB += TK;
  }

  // epilogue: rank-1 mean correction, scale, mirrored symmetric store
  const float* mb = mean + (size_t)b * Cn;
  const int lane = t & 31;
  const int hl   = lane >> 4;
  const int nn   = lane & 15;
  float* sg = sigma + (size_t)b * Cn * Cn;
#pragma unroll
  for (int mi = 0; mi < 4; ++mi)
#pragma unroll
    for (int ni = 0; ni < 4; ++ni) {
      const int   ccol = d0 + wn * 64 + ni * 16 + nn;
      const float mu_d = mb[ccol] * (float)Nn;
#pragma unroll
      for (int r = 0; r < 8; ++r) {
        const int   crow = c0 + wm * 64 + mi * 16 + r + 8 * hl;
        const float val  = (acc[mi][ni][r] - mb[crow] * mu_d) * SIGMA_SCALE;
        sg[(size_t)crow * Cn + ccol] = val;
        sg[(size_t)ccol * Cn + crow] = val;   // mirror (r-contig per lane)
      }
    }
}

// ------------------------- kernel 3: row softmax ---------------------------
__global__ __launch_bounds__(256)
void cam_softmax(float* __restrict__ sigma) {
  const int row = blockIdx.x;                  // B*C rows of width C=512
  float* p = sigma + (size_t)row * Cn;
  const int t = threadIdx.x;
  const float v0 = p[t], v1 = p[t + 256];
  float m = fmaxf(v0, v1);
#pragma unroll
  for (int off = 16; off; off >>= 1) m = fmaxf(m, __shfl_xor(m, off, 32));
  __shared__ float red[8];
  if ((t & 31) == 0) red[t >> 5] = m;
  __syncthreads();
  m = red[0];
#pragma unroll
  for (int i = 1; i < 8; ++i) m = fmaxf(m, red[i]);
  const float e0 = __expf(v0 - m), e1 = __expf(v1 - m);
  float s = e0 + e1;
#pragma unroll
  for (int off = 16; off; off >>= 1) s += __shfl_xor(s, off, 32);
  __syncthreads();
  if ((t & 31) == 0) red[t >> 5] = s;
  __syncthreads();
  s = 0.f;
#pragma unroll
  for (int i = 0; i < 8; ++i) s += red[i];
  const float inv = 1.0f / s;
  p[t]       = e0 * inv;
  p[t + 256] = e1 * inv;
}

// ----------- kernel 4: out = attn @ Xf, d_out = gamma*out + x --------------
// Per batch: M=C, N=Nn, K=C. B tile (Xf rows k, cols n) transposed into LDS
// as [n][k] so B-fragment reads are K-contiguous.
__global__ __launch_bounds__(128)
void cam_out_gemm(const float* __restrict__ attn, const float* __restrict__ x,
                  const float* __restrict__ gamma, float* __restrict__ out) {
  constexpr int TM = 128, TN = 128, TK = 64, LDK = TK + 8;
  __shared__ bf16_t sA[TM * LDK];   // attn tile [c][k]
  __shared__ bf16_t sB[TN * LDK];   // x tile    [n][k]

  const int b  = blockIdx.z;
  const int c0 = blockIdx.y * TM;
  const int n0 = blockIdx.x * TN;
  const float* ab = attn + (size_t)b * Cn * Cn;
  const float* xb = x    + (size_t)b * Cn * Nn;

  const int t    = threadIdx.x;
  const int wave = t >> 5;
  const int wm   = wave >> 1;
  const int wn   = wave & 1;

  // A-tile bases (rows step 8 per iter)
  const int rbase = t >> 4;          // 0..7
  const int cbase = (t & 15) << 2;   // 0..60
  const float* pA = ab + (size_t)(c0 + rbase) * Cn + cbase;
  bf16_t* dA = &sA[rbase * LDK + cbase];
  // B-tile bases (transposed store; k rows step 4 per iter)
  const int krb   = t >> 5;          // 0..3
  const int cb4   = (t & 31) << 2;   // 0..124
  const float* pB = xb + (size_t)krb * Nn + n0 + cb4;
  bf16_t* dB = &sB[cb4 * LDK + krb];

  v8f acc[4][4] = {};

  for (int k0 = 0; k0 < Cn; k0 += TK) {
    const bool pf = (k0 + TK < Cn);
#pragma unroll
    for (int i = 0; i < 16; ++i) {           // attn tile: 128 rows x 64 k
      const float4 v = *(const float4*)(pA + i * 8 * Cn);
      bf16_t* da = dA + i * 8 * LDK;
      da[0] = (bf16_t)v.x; da[1] = (bf16_t)v.y;
      da[2] = (bf16_t)v.z; da[3] = (bf16_t)v.w;
      if (pf) __builtin_prefetch((const void*)(pA + i * 8 * Cn + TK), 0, 1);
    }
#pragma unroll
    for (int i = 0; i < 16; ++i) {           // x tile: 64 k-rows x 128 n
      const float4 v = *(const float4*)(pB + i * 4 * Nn);
      bf16_t* db = dB + i * 4;
      db[0 * LDK] = (bf16_t)v.x;
      db[1 * LDK] = (bf16_t)v.y;
      db[2 * LDK] = (bf16_t)v.z;
      db[3 * LDK] = (bf16_t)v.w;
      if (pf) __builtin_prefetch((const void*)(pB + i * 4 * Nn + TK * Nn), 0, 1);
    }
    __syncthreads();

#pragma unroll
    for (int kk = 0; kk < TK; kk += 32) {
      v16bf af[4], bfr[4];
#pragma unroll
      for (int mi = 0; mi < 4; ++mi)
        af[mi] = frag_a_ld(&sA[(wm * 64 + mi * 16) * LDK + kk], LDK);
#pragma unroll
      for (int ni = 0; ni < 4; ++ni)
        bfr[ni] = frag_b_ld(&sB[(wn * 64 + ni * 16) * LDK + kk], LDK);
#pragma unroll
      for (int mi = 0; mi < 4; ++mi)
#pragma unroll
        for (int ni = 0; ni < 4; ++ni)
          acc[mi][ni] = __builtin_amdgcn_wmma_f32_16x16x32_bf16(
              false, af[mi], false, bfr[ni], (short)0, acc[mi][ni], false, false);
    }
    __syncthreads();
    pA += TK;
    pB += (size_t)TK * Nn;
  }

  const float g  = gamma[0];
  const int lane = t & 31;
  const int hl   = lane >> 4;
  const int nn   = lane & 15;
  float* ob = out + (size_t)b * Cn * Nn;
  const float* xrd = xb;
#pragma unroll
  for (int mi = 0; mi < 4; ++mi)
#pragma unroll
    for (int ni = 0; ni < 4; ++ni) {
      const int col = n0 + wn * 64 + ni * 16 + nn;
#pragma unroll
      for (int r = 0; r < 8; ++r) {
        const int row   = c0 + wm * 64 + mi * 16 + r + 8 * hl;
        const size_t of = (size_t)row * Nn + col;
        ob[of] = fmaf(g, acc[mi][ni][r], xrd[of]);
      }
    }
}

// ------------------------------- launcher ----------------------------------
extern "C" void kernel_launch(void* const* d_in, const int* in_sizes, int n_in,
                              void* d_out, int out_size, void* d_ws, size_t ws_size,
                              hipStream_t stream) {
  const float* x     = (const float*)d_in[0];
  const float* gamma = (const float*)d_in[1];
  float*       out   = (float*)d_out;

  // workspace: [0,64KB) row means (B*C f32 = 32KB used), [64KB, ..) sigma/attn
  float* mean  = (float*)d_ws;
  float* sigma = (float*)((char*)d_ws + 64 * 1024);   // B*C*C f32 = 16 MiB

  constexpr int TRI_TILES = (Cn / 128) * (Cn / 128 + 1) / 2;   // 10
  cam_row_mean  <<<dim3(Bn * Cn),                dim3(256), 0, stream>>>(x, mean);
  cam_sigma_gemm<<<dim3(TRI_TILES, 1, Bn),       dim3(128), 0, stream>>>(x, mean, sigma);
  cam_softmax   <<<dim3(Bn * Cn),                dim3(256), 0, stream>>>(sigma);
  cam_out_gemm  <<<dim3(Nn / 128, Cn / 128, Bn), dim3(128), 0, stream>>>(sigma, x, gamma, out);
}
